// WSAttention_56495999811809
// MI455X (gfx1250) — matrix operand order repaired and checked by reading, not compiled, not run on hardware
//
#include <hip/hip_runtime.h>
#include <hip/hip_bf16.h>

typedef __bf16 bf16_t;
typedef bf16_t v16bf __attribute__((ext_vector_type(16)));
typedef bf16_t v8bf  __attribute__((ext_vector_type(8)));
typedef float  v8f   __attribute__((ext_vector_type(8)));
typedef unsigned int v4u __attribute__((ext_vector_type(4)));
typedef int v4i __attribute__((ext_vector_type(4)));
typedef int v8i __attribute__((ext_vector_type(8)));

static __device__ __forceinline__ v16bf pack16(v8bf lo, v8bf hi) {
  return __builtin_shufflevector(lo, hi, 0,1,2,3,4,5,6,7,8,9,10,11,12,13,14,15);
}

static __device__ __forceinline__ v8f wmma_bf16(v16bf a, v16bf b, v8f c) {
  // D = A(16x32 bf16) * B(32x16 bf16) + C(16x16 f32)
  return __builtin_amdgcn_wmma_f32_16x16x32_bf16(false, a, false, b, (short)0, c,
                                                 false, false);
}

// ---------------------------------------------------------------------------
// TDM: async-load one K tile [32 keys x 128 dh] of bf16 (row stride 128) from
// global into LDS with 8-bf16 row padding (pitch 136). 2-D descriptor per
// CDNA5 ISA. Issued once per wave; TENSORcnt tracked.
// ---------------------------------------------------------------------------
static __device__ __forceinline__ void tdm_load_k_tile(const bf16_t* gptr,
                                                       unsigned lds_off) {
  unsigned long long ga = (unsigned long long)(size_t)gptr;
  v4u g0;
  g0[0] = 1u;                                    // count=1, user mode
  g0[1] = lds_off;                               // lds_addr (bytes)
  g0[2] = (unsigned)(ga & 0xFFFFFFFFu);          // global_addr[31:0]
  g0[3] = (unsigned)((ga >> 32) & 0x01FFFFFFu)   // global_addr[56:32]
        | (2u << 30);                            // type=2 ("image")
  v8i g1;
  g1[0] = (int)((1u << 16)    // data_size = 2 bytes
              | (1u << 20)    // pad_enable
              | (5u << 22)    // pad_interval: 64 DWORDs (= one 256B row)
              | (3u << 25));  // pad_amount: 4 DWORDs (= 8 bf16) -> pitch 136
  g1[1] = (int)(128u  << 16); // tensor_dim0 = 128      ([79:48] lo16)
  g1[2] = (int)(2048u << 16); // dim0 hi=0, tensor_dim1 = 2048 lo16
  g1[3] = (int)(128u  << 16); // dim1 hi=0, tile_dim0 = 128
  g1[4] = 32;                 // tile_dim1 = 32, tile_dim2 = 0
  g1[5] = 128;                // tensor_dim0_stride lo32 = 128 elements
  g1[6] = 0;                  // stride hi, tensor_dim1_stride lo
  g1[7] = 0;
  v4i z4 = {0, 0, 0, 0};
#if defined(__clang_major__) && (__clang_major__ >= 23)
  v8i z8 = {0, 0, 0, 0, 0, 0, 0, 0};
  __builtin_amdgcn_tensor_load_to_lds(g0, g1, z4, z4, z8, 0);
#else
  __builtin_amdgcn_tensor_load_to_lds(g0, g1, z4, z4, 0);
#endif
}

// ---------------------------------------------------------------------------
// Tiled GEMM: out[M,N] = A[M,K] * W[K,N] + bias[N]
//   ABF16=false : A is fp32 (converted to bf16 while staging to LDS)
//   ABF16=true  : A is bf16 already
//   EPI=0 : store fp32 to outF[row*N+col]
//   EPI=1 : QKV scatter -> q/k/v bf16 in [B*H, S, 128] layout
// Block tile: BM=64, BN=128, BK=32; 8 waves, each 32x32 of output.
// ---------------------------------------------------------------------------
template<bool ABF16, int EPI>
__global__ __launch_bounds__(256) void gemm_wmma_kernel(
    const void* __restrict__ Av, const float* __restrict__ W,
    const float* __restrict__ bias, float* __restrict__ outF,
    bf16_t* __restrict__ q_o, bf16_t* __restrict__ k_o, bf16_t* __restrict__ v_o,
    int M, int N, int K)
{
  constexpr int BM = 64, BN = 128, BK = 32, PA = 40, PB = 40;
  __shared__ __align__(16) bf16_t Asub[BM * PA];
  __shared__ __align__(16) bf16_t Bt[BN * PB];

  const int tid  = threadIdx.x;
  const int wave = tid >> 5, lane = tid & 31;
  const int lhi  = lane >> 4, llo = lane & 15;
  const int wm   = wave >> 2, wn = wave & 3;
  const int m0   = blockIdx.y * BM, n0 = blockIdx.x * BN;

  const float*  Af = (const float*)Av;
  const bf16_t* Ab = (const bf16_t*)Av;

  v8f acc[2][2] = {};

  for (int k0 = 0; k0 < K; k0 += BK) {
    // ---- stage A tile [64 x 32] as bf16 (row-major, pitch PA) ----
    if constexpr (!ABF16) {
      #pragma unroll
      for (int i = 0; i < 2; ++i) {
        int idx = tid + i * 256;               // 512 float4 chunks
        int r = idx >> 3, c4 = idx & 7;
        float4 f = *(const float4*)(Af + (size_t)(m0 + r) * K + k0 + c4 * 4);
        bf16_t* dst = &Asub[r * PA + c4 * 4];
        dst[0] = (bf16_t)f.x; dst[1] = (bf16_t)f.y;
        dst[2] = (bf16_t)f.z; dst[3] = (bf16_t)f.w;
      }
    } else {
      int r = tid >> 2, c8 = tid & 3;          // 256 chunks of 8 bf16
      *(uint4*)(&Asub[r * PA + c8 * 8]) =
          *(const uint4*)(Ab + (size_t)(m0 + r) * K + k0 + c8 * 8);
    }
    // ---- stage B tile [32 x 128] transposed: Bt[n][k] ----
    #pragma unroll
    for (int i = 0; i < 4; ++i) {
      int idx = tid + i * 256;                 // 1024 float4 chunks
      int kk = idx >> 5, c4 = idx & 31;
      float4 f = *(const float4*)(W + (size_t)(k0 + kk) * N + n0 + c4 * 4);
      int nb = c4 * 4;
      Bt[(nb + 0) * PB + kk] = (bf16_t)f.x;
      Bt[(nb + 1) * PB + kk] = (bf16_t)f.y;
      Bt[(nb + 2) * PB + kk] = (bf16_t)f.z;
      Bt[(nb + 3) * PB + kk] = (bf16_t)f.w;
    }
    if (k0 + BK < K) {
      if constexpr (ABF16)
        __builtin_prefetch(Ab + (size_t)(m0 + (tid >> 2)) * K + k0 + BK, 0, 1);
      else
        __builtin_prefetch(Af + (size_t)(m0 + (tid >> 3)) * K + k0 + BK, 0, 1);
      __builtin_prefetch(W + (size_t)(k0 + BK + (tid >> 5)) * N + n0, 0, 1);
    }
    __syncthreads();

    // ---- fragments + 4 WMMAs per wave ----
    const bf16_t* pa0 = &Asub[(wm * 32 + llo) * PA];
    const bf16_t* pa1 = &Asub[(wm * 32 + 16 + llo) * PA];
    v16bf a0 = pack16(*(const v8bf*)(pa0 + 8 * lhi), *(const v8bf*)(pa0 + 16 + 8 * lhi));
    v16bf a1 = pack16(*(const v8bf*)(pa1 + 8 * lhi), *(const v8bf*)(pa1 + 16 + 8 * lhi));
    const bf16_t* pb0 = &Bt[(wn * 32 + llo) * PB];
    const bf16_t* pb1 = &Bt[(wn * 32 + 16 + llo) * PB];
    v16bf b0 = pack16(*(const v8bf*)(pb0 + 8 * lhi), *(const v8bf*)(pb0 + 16 + 8 * lhi));
    v16bf b1 = pack16(*(const v8bf*)(pb1 + 8 * lhi), *(const v8bf*)(pb1 + 16 + 8 * lhi));

    acc[0][0] = wmma_bf16(a0, b0, acc[0][0]);
    acc[0][1] = wmma_bf16(a0, b1, acc[0][1]);
    acc[1][0] = wmma_bf16(a1, b0, acc[1][0]);
    acc[1][1] = wmma_bf16(a1, b1, acc[1][1]);
    __syncthreads();
  }

  // ---- epilogue ----
  #pragma unroll
  for (int sm = 0; sm < 2; ++sm)
    #pragma unroll
    for (int sn = 0; sn < 2; ++sn)
      #pragma unroll
      for (int r = 0; r < 8; ++r) {
        int row = m0 + wm * 32 + sm * 16 + lhi * 8 + r;
        int col = n0 + wn * 32 + sn * 16 + llo;
        float val = acc[sm][sn][r] + bias[col];
        if constexpr (EPI == 0) {
          outF[(size_t)row * N + col] = val;
        } else {
          int sel = col >> 11;                  // 0=q 1=k 2=v
          int rem = col & 2047;
          int h = rem >> 7, dh = rem & 127;
          int bb = row >> 11, s = row & 2047;   // S = 2048
          size_t dst = ((size_t)(bb * 16 + h) * 2048 + s) * 128 + dh;
          bf16_t bv = (bf16_t)val;
          if (sel == 0)      q_o[dst] = bv;
          else if (sel == 1) k_o[dst] = bv;
          else               v_o[dst] = bv;
        }
      }
}

// ---------------------------------------------------------------------------
// Flash attention: one block per (b, h, 128-row q tile). 8 waves x 16 q rows.
// K tiles double-buffered via TDM (tensor_load_to_lds, TENSORcnt); V staged
// transposed by the vector pipe. Causal early termination over 32-key steps.
// ---------------------------------------------------------------------------
__global__ __launch_bounds__(256) void flash_attn_kernel(
    const bf16_t* __restrict__ qb, const bf16_t* __restrict__ kb,
    const bf16_t* __restrict__ vb, bf16_t* __restrict__ concat)
{
  constexpr int S = 2048, DH = 128, PK = 136, PV = 40, PP = 40;
  const float SCALE = 0.08838834764831845f;   // 1/sqrt(128)

  __shared__ __align__(16) bf16_t Ksub[2][32 * PK]; // double-buffered K tiles
  __shared__ __align__(16) bf16_t Vt[DH * PV];      // V tile, transposed [d][key]
  __shared__ __align__(16) bf16_t Pw[8 * 16 * PP];  // per-wave P transpose area

  const int tid  = threadIdx.x;
  const int wave = tid >> 5, lane = tid & 31;
  const int lhi  = lane >> 4, llo = lane & 15;
  const int qblk = blockIdx.x & 15, bh = blockIdx.x >> 4;
  const int b = bh >> 4, h = bh & 15;
  const int q0 = qblk * 128;

  const bf16_t* Qh = qb + (size_t)bh * S * DH;
  const bf16_t* Kh = kb + (size_t)bh * S * DH;
  const bf16_t* Vh = vb + (size_t)bh * S * DH;

  // Q A-fragments (row = llo, 4 chunks of K=32 over Dh=128), kept in VGPRs
  v16bf qfrag[4];
  {
    const bf16_t* qp = Qh + (size_t)(q0 + wave * 16 + llo) * DH;
    #pragma unroll
    for (int c = 0; c < 4; ++c) {
      const bf16_t* p = qp + 32 * c + 8 * lhi;
      qfrag[c] = pack16(*(const v8bf*)p, *(const v8bf*)(p + 16));
    }
  }

  v8f o[8] = {};
  float m_run[8], l_run[8];
  #pragma unroll
  for (int r = 0; r < 8; ++r) { m_run[r] = -1e30f; l_run[r] = 0.0f; }

  const unsigned koff[2] = { (unsigned)(size_t)&Ksub[0][0],
                             (unsigned)(size_t)&Ksub[1][0] };
  const int steps = (q0 + 128) >> 5;          // causal: keys <= q0+127

  // prologue: async-load tile 0
  if (wave == 0) tdm_load_k_tile(Kh, koff[0]);

  for (int st = 0; st < steps; ++st) {
    const int kb0 = st * 32;

    // issue TDM for the next K tile (overwrites buffer freed at st-1's barrier)
    if (wave == 0 && st + 1 < steps)
      tdm_load_k_tile(Kh + (size_t)(st + 1) * 32 * DH, koff[(st + 1) & 1]);

    // stage V tile transposed: Vt[d][key]   (all 256 threads)
    #pragma unroll
    for (int i = 0; i < 2; ++i) {
      int idx = tid + i * 256;
      int key = idx >> 4, c8 = idx & 15;
      union { uint4 u; bf16_t e[8]; } cvt;
      cvt.u = *(const uint4*)(Vh + (size_t)(kb0 + key) * DH + c8 * 8);
      #pragma unroll
      for (int j = 0; j < 8; ++j) Vt[(c8 * 8 + j) * PV + key] = cvt.e[j];
    }

    // current K tile must be resident (TENSORcnt is in-order per wave)
    if (wave == 0) {
      if (st + 1 < steps) __builtin_amdgcn_s_wait_tensorcnt(1);
      else                __builtin_amdgcn_s_wait_tensorcnt(0);
    }
    __syncthreads();

    const bf16_t* Kt = &Ksub[st & 1][0];

    // scores: two 16x16 tiles (keys kb0..+15, kb0+16..+31), K-dim = Dh
    v8f c0 = {}, c1 = {};
    #pragma unroll
    for (int c = 0; c < 4; ++c) {
      const bf16_t* p0 = &Kt[llo * PK + 32 * c + 8 * lhi];
      v16bf bk0 = pack16(*(const v8bf*)p0, *(const v8bf*)(p0 + 16));
      c0 = wmma_bf16(qfrag[c], bk0, c0);
      const bf16_t* p1 = &Kt[(16 + llo) * PK + 32 * c + 8 * lhi];
      v16bf bk1 = pack16(*(const v8bf*)p1, *(const v8bf*)(p1 + 16));
      c1 = wmma_bf16(qfrag[c], bk1, c1);
    }

    // online softmax (row groups = 16 lanes sharing lane>>4)
    float alpha[8];
    bf16_t* pwrow = &Pw[wave * 16 * PP];
    #pragma unroll
    for (int r = 0; r < 8; ++r) {
      const int rowg = q0 + wave * 16 + lhi * 8 + r;
      float v0 = c0[r] * SCALE; if (kb0 + llo      > rowg) v0 = -1e30f;
      float v1 = c1[r] * SCALE; if (kb0 + 16 + llo > rowg) v1 = -1e30f;
      float mx = fmaxf(v0, v1);
      mx = fmaxf(mx, __shfl_xor(mx, 1, 32));
      mx = fmaxf(mx, __shfl_xor(mx, 2, 32));
      mx = fmaxf(mx, __shfl_xor(mx, 4, 32));
      mx = fmaxf(mx, __shfl_xor(mx, 8, 32));
      float mnew = fmaxf(m_run[r], mx);
      float al = __expf(m_run[r] - mnew);
      float p0 = __expf(v0 - mnew);
      float p1 = __expf(v1 - mnew);
      float sum = p0 + p1;
      sum += __shfl_xor(sum, 1, 32);
      sum += __shfl_xor(sum, 2, 32);
      sum += __shfl_xor(sum, 4, 32);
      sum += __shfl_xor(sum, 8, 32);
      l_run[r] = l_run[r] * al + sum;
      m_run[r] = mnew;
      alpha[r] = al;
      // transpose P through wave-private LDS (C layout -> A layout)
      pwrow[(lhi * 8 + r) * PP + llo]      = (bf16_t)p0;
      pwrow[(lhi * 8 + r) * PP + 16 + llo] = (bf16_t)p1;
    }
    asm volatile("s_wait_dscnt 0" ::: "memory");
    const bf16_t* pp = pwrow + llo * PP + 8 * lhi;
    v16bf pfrag = pack16(*(const v8bf*)pp, *(const v8bf*)(pp + 16));

    // rescale accumulators, then P*V (8 d-chunks of 16)
    #pragma unroll
    for (int n = 0; n < 8; ++n)
      #pragma unroll
      for (int r = 0; r < 8; ++r) o[n][r] *= alpha[r];
    #pragma unroll
    for (int n = 0; n < 8; ++n) {
      const bf16_t* vp = &Vt[(16 * n + llo) * PV + 8 * lhi];
      v16bf bv = pack16(*(const v8bf*)vp, *(const v8bf*)(vp + 16));
      o[n] = wmma_bf16(pfrag, bv, o[n]);
    }
    __syncthreads();
  }

  // epilogue: normalize and write bf16 concat [B, S, D]
  #pragma unroll
  for (int n = 0; n < 8; ++n)
    #pragma unroll
    for (int r = 0; r < 8; ++r) {
      int row = q0 + wave * 16 + lhi * 8 + r;
      int d = 16 * n + llo;
      float val = o[n][r] / l_run[r];
      concat[((size_t)b * S + row) * 2048 + (size_t)h * 128 + d] = (bf16_t)val;
    }
}

// ---------------------------------------------------------------------------
extern "C" void kernel_launch(void* const* d_in, const int* in_sizes, int n_in,
                              void* d_out, int out_size, void* d_ws, size_t ws_size,
                              hipStream_t stream) {
  (void)in_sizes; (void)n_in; (void)out_size; (void)ws_size;
  const float* x     = (const float*)d_in[0];  // [2,2048,2048]
  const float* W_qkv = (const float*)d_in[1];  // [2048,6144]
  const float* b_qkv = (const float*)d_in[2];  // [6144]
  const float* W_out = (const float*)d_in[3];  // [2048,2048]
  const float* b_out = (const float*)d_in[4];  // [2048]
  float* out = (float*)d_out;                  // [2,2048,2048] fp32

  char* ws = (char*)d_ws;
  const size_t SZ = (size_t)32 * 2048 * 128 * sizeof(bf16_t); // 16 MiB each
  bf16_t* qb     = (bf16_t*)(ws);
  bf16_t* kbuf   = (bf16_t*)(ws + SZ);
  bf16_t* vbuf   = (bf16_t*)(ws + 2 * SZ);
  bf16_t* concat = (bf16_t*)(ws + 3 * SZ);

  // 1) QKV projection: [4096,2048] x [2048,6144] + bias -> q/k/v bf16 heads
  dim3 g1(6144 / 128, 4096 / 64);
  gemm_wmma_kernel<false, 1><<<g1, 256, 0, stream>>>(
      (const void*)x, W_qkv, b_qkv, nullptr, qb, kbuf, vbuf, 4096, 6144, 2048);

  // 2) causal flash attention (TDM-fed K tiles) -> concat bf16 [B,S,D]
  flash_attn_kernel<<<2 * 16 * 16, 256, 0, stream>>>(qb, kbuf, vbuf, concat);

  // 3) output projection: [4096,2048] x [2048,2048] + bias -> fp32 out
  dim3 g2(2048 / 128, 4096 / 64);
  gemm_wmma_kernel<true, 0><<<g2, 256, 0, stream>>>(
      (const void*)concat, W_out, b_out, out, nullptr, nullptr, nullptr,
      4096, 2048, 2048);
}
